// Model_51754355916897
// MI455X (gfx1250) — matrix operand order repaired and checked by reading, not compiled, not run on hardware
//
#include <hip/hip_runtime.h>
#include <hip/hip_bf16.h>
#include <math.h>

// ---------------------------------------------------------------------------
// MoE FFN (top-2 of 8 experts), B=256, L=64, D=1024, H=4096.
// v2: M=64 tokens/block (one batch row) so each expert's W1/W2 slab is read
// exactly ONCE per block -> ~8 GB total weight traffic (vs 33 GB at M=16),
// balancing L2 bandwidth against ~110us of dense-bf16 WMMA work. Uses the
// full CDNA5 320KB-LDS WGP: 128KB x-row + 32KB h-chunk = 160KB, 16 waves.
// x row staged via global_load_async_to_lds_b128 (ASYNCcnt path).
// Workspace: gates 8KB + W1t 67.1MB + W2t 67.1MB + x_bf16 33.6MB ~= 168MB.
// ---------------------------------------------------------------------------

#define B_DIM 256
#define L_DIM 64
#define D_DIM 1024
#define H_DIM 4096
#define E_NUM 8
#define HC    256   // h-chunk staged through LDS per iteration

typedef __attribute__((ext_vector_type(16))) __bf16 v16bf;
typedef __attribute__((ext_vector_type(8)))  float  v8f;

union FragBF {
    uint4 q[2];
    v16bf v;
};

__device__ __forceinline__ unsigned short f2bf(float f) {
    unsigned u = __float_as_uint(f);
    u += 0x7FFFu + ((u >> 16) & 1u);      // round-to-nearest-even
    return (unsigned short)(u >> 16);
}

__device__ __forceinline__ float gelu_exact(float x) {
    return 0.5f * x * (1.0f + erff(x * 0.70710678118654752440f));
}

// ---------------------------------------------------------------------------
// Kernel 1: router gates. One thread per batch row.
// ---------------------------------------------------------------------------
__global__ void gates_kernel(const float* __restrict__ logits,
                             const int* __restrict__ masks,
                             float* __restrict__ gates) {
    int b = threadIdx.x;              // blockDim.x == B_DIM
    float p[E_NUM];
    float mx = -3.402823466e38f;
#pragma unroll
    for (int e = 0; e < E_NUM; ++e) {
        p[e] = logits[b * E_NUM + e];
        mx = fmaxf(mx, p[e]);
    }
    float s = 0.f;
#pragma unroll
    for (int e = 0; e < E_NUM; ++e) { p[e] = expf(p[e] - mx); s += p[e]; }
    float inv = 1.0f / s;
#pragma unroll
    for (int e = 0; e < E_NUM; ++e) {
        p[e] *= inv;
        if (masks[b * E_NUM + e] != 1) p[e] = 0.f;
    }
    int i1 = -1; float g1 = -1.f;
#pragma unroll
    for (int e = 0; e < E_NUM; ++e) if (p[e] > g1) { g1 = p[e]; i1 = e; }
    int i2 = -1; float g2 = -1.f;
#pragma unroll
    for (int e = 0; e < E_NUM; ++e) if (e != i1 && p[e] > g2) { g2 = p[e]; i2 = e; }
    float den = 1.0f / (g1 + g2 + 1e-9f);
#pragma unroll
    for (int e = 0; e < E_NUM; ++e)
        gates[b * E_NUM + e] = (e == i1 ? g1 : (e == i2 ? g2 : 0.f)) * den;
}

// ---------------------------------------------------------------------------
// Kernel 2: fp32 [R][C] -> bf16 [C][R] per expert slice (32x32 LDS tiles).
// W1[e][D][H] -> W1t[e][H][D];  W2[e][H][D] -> W2t[e][D][H]
// ---------------------------------------------------------------------------
__global__ void transpose_to_bf16(const float* __restrict__ src,
                                  unsigned short* __restrict__ dst,
                                  int R, int C) {
    __shared__ float tile[32][33];
    const size_t esz = (size_t)R * C;
    const float* s = src + (size_t)blockIdx.z * esz;
    unsigned short* d = dst + (size_t)blockIdx.z * esz;
    int c0 = blockIdx.x * 32, r0 = blockIdx.y * 32;
#pragma unroll
    for (int i = threadIdx.y; i < 32; i += 8)
        tile[i][threadIdx.x] = s[(size_t)(r0 + i) * C + c0 + threadIdx.x];
    __syncthreads();
#pragma unroll
    for (int i = threadIdx.y; i < 32; i += 8)
        d[(size_t)(c0 + i) * R + r0 + threadIdx.x] = f2bf(tile[threadIdx.x][i]);
}

// ---------------------------------------------------------------------------
// Kernel 3: x fp32 -> bf16 flat copy (enables async-to-LDS staging).
// ---------------------------------------------------------------------------
__global__ void convert_x_bf16(const float* __restrict__ x,
                               unsigned short* __restrict__ xb, int n4) {
    int i = blockIdx.x * blockDim.x + threadIdx.x;
    if (i >= n4) return;
    float4 v = reinterpret_cast<const float4*>(x)[i];
    uint2 pk;
    pk.x = (unsigned)f2bf(v.x) | ((unsigned)f2bf(v.y) << 16);
    pk.y = (unsigned)f2bf(v.z) | ((unsigned)f2bf(v.w) << 16);
    reinterpret_cast<uint2*>(xb)[i] = pk;
}

// ---------------------------------------------------------------------------
// Kernel 4: fused expert FFN.
// grid = (B); block = 512 threads (16 waves, wave32).
// Block owns one batch row: 64 tokens x full D=1024 output.
//   GEMM1: wave w -> h-tile w of the 256-wide chunk, 4 token-tiles (B reused 4x)
//   GEMM2: wave w -> d-tiles 4w..4w+3, 4 token-tiles (B reused 4x, slab read 1x)
// ---------------------------------------------------------------------------
__global__ __launch_bounds__(512) void moe_ffn_kernel(
    const unsigned short* __restrict__ xb,    // [B][L][D] bf16
    const float* __restrict__ gates,          // [B][E]
    const unsigned short* __restrict__ W1t,   // [E][H][D] bf16
    const unsigned short* __restrict__ W2t,   // [E][D][H] bf16
    const float* __restrict__ b2,             // [E][D] f32
    unsigned short* __restrict__ out)         // [B][L][D] bf16
{
    __shared__ unsigned short xA[L_DIM * D_DIM];   // 128 KB: full x row, bf16
    __shared__ unsigned short hbuf[L_DIM * HC];    //  32 KB: gelu(h)*g chunk

    const int b   = blockIdx.x;
    const int tid = threadIdx.x;
    const int w     = tid >> 5;       // wave 0..15
    const int lane  = tid & 31;
    const int lhalf = lane >> 4;      // selects K sub-range of the fragment
    const int l15   = lane & 15;

    // ---- async-stage x row: 64x1024 bf16 = 128 KB, 16B chunks ----
    {
        const unsigned long long gbase =
            (unsigned long long)(uintptr_t)(xb + (size_t)b * L_DIM * D_DIM);
        const unsigned lbase = (unsigned)(uintptr_t)(&xA[0]);
#pragma unroll
        for (int it = 0; it < 16; ++it) {
            const int c = it * 512 + tid;                  // 8192 chunks total
            unsigned long long ga = gbase + (unsigned long long)c * 16u;
            unsigned la = lbase + (unsigned)c * 16u;
            asm volatile("global_load_async_to_lds_b128 %0, %1, off"
                         :: "v"(la), "v"(ga) : "memory");
        }
        asm volatile("s_wait_asynccnt 0x0" ::: "memory");
    }
    __syncthreads();

    v8f o[16] = {};   // [token-tile 0..3][d-tile 0..3], 64 tok x 64 d per wave

    for (int e = 0; e < E_NUM; ++e) {
        const float g = gates[b * E_NUM + e];   // uniform across block
        if (g == 0.0f) continue;                // skip dead experts (~6 of 8)
        const unsigned short* W1e = W1t + (size_t)e * H_DIM * D_DIM;
        const unsigned short* W2e = W2t + (size_t)e * D_DIM * H_DIM;

        for (int hc = 0; hc < H_DIM; hc += HC) {
            // ---- GEMM1: h-tile w: htile[64 x 16] = x[64x1024] @ W1[1024x16]
            v8f c[4] = {};
            const unsigned short* w1p =
                W1e + (size_t)(hc + w * 16 + l15) * D_DIM + lhalf * 8;
            for (int k = 0; k < D_DIM; k += 32) {
                const uint4* bp = reinterpret_cast<const uint4*>(w1p + k);
                FragBF bb;
                bb.q[0] = bp[0]; bb.q[1] = bp[2];
                __builtin_prefetch(reinterpret_cast<const char*>(bp) + 256, 0, 0);
#pragma unroll
                for (int tt = 0; tt < 4; ++tt) {
                    FragBF a;
                    const uint4* ap = reinterpret_cast<const uint4*>(
                        &xA[(tt * 16 + l15) * D_DIM + k + lhalf * 8]);
                    a.q[0] = ap[0];          // K kb..kb+7
                    a.q[1] = ap[2];          // K kb+16..kb+23
                    c[tt] = __builtin_amdgcn_wmma_f32_16x16x32_bf16(
                                false, a.v, false, bb.v, (short)0, c[tt],
                                false, false);
                }
            }
            // exact GELU, gate folded in (g*(h@W2) == (g*h)@W2) -> LDS
#pragma unroll
            for (int tt = 0; tt < 4; ++tt)
#pragma unroll
                for (int r = 0; r < 8; ++r) {
                    int m = tt * 16 + r + lhalf * 8;
                    hbuf[m * HC + w * 16 + l15] = f2bf(gelu_exact(c[tt][r]) * g);
                }
            __syncthreads();

            // ---- GEMM2: o[64 x 64] += htile[64 x 256] @ W2[256 x 64] ----
            for (int k = 0; k < HC; k += 32) {
                FragBF a[4];
#pragma unroll
                for (int tt = 0; tt < 4; ++tt) {
                    const uint4* ap = reinterpret_cast<const uint4*>(
                        &hbuf[(tt * 16 + l15) * HC + k + lhalf * 8]);
                    a[tt].q[0] = ap[0];
                    a[tt].q[1] = ap[2];
                }
#pragma unroll
                for (int dt = 0; dt < 4; ++dt) {
                    const int dcol = (w * 4 + dt) * 16 + l15;
                    const uint4* bp = reinterpret_cast<const uint4*>(
                        W2e + (size_t)dcol * H_DIM + hc + k + lhalf * 8);
                    FragBF bb;
                    bb.q[0] = bp[0]; bb.q[1] = bp[2];
#pragma unroll
                    for (int tt = 0; tt < 4; ++tt)
                        o[tt * 4 + dt] = __builtin_amdgcn_wmma_f32_16x16x32_bf16(
                            false, a[tt].v, false, bb.v, (short)0,
                            o[tt * 4 + dt], false, false);
                }
            }
            __syncthreads();   // hbuf reused next chunk
        }
    }

    // ---- epilogue: gate-weighted bias, store bf16 ----
#pragma unroll
    for (int dt = 0; dt < 4; ++dt) {
        const int col = (w * 4 + dt) * 16 + l15;
        float bias = 0.f;
#pragma unroll
        for (int e = 0; e < E_NUM; ++e)
            bias += gates[b * E_NUM + e] * b2[e * D_DIM + col];
#pragma unroll
        for (int tt = 0; tt < 4; ++tt)
#pragma unroll
            for (int r = 0; r < 8; ++r) {
                int row = tt * 16 + r + lhalf * 8;
                out[(size_t)(b * L_DIM + row) * D_DIM + col] =
                    f2bf(o[tt * 4 + dt][r] + bias);
            }
    }
}

// ---------------------------------------------------------------------------
extern "C" void kernel_launch(void* const* d_in, const int* in_sizes, int n_in,
                              void* d_out, int out_size, void* d_ws, size_t ws_size,
                              hipStream_t stream) {
    const float* x      = (const float*)d_in[0];   // [B,L,D]
    const float* logits = (const float*)d_in[1];   // [B,E]
    const int*   masks  = (const int*)d_in[2];     // [B,E]
    const float* W1     = (const float*)d_in[3];   // [E,D,H]
    const float* W2     = (const float*)d_in[4];   // [E,H,D]
    const float* b2     = (const float*)d_in[5];   // [E,D]
    unsigned short* out = (unsigned short*)d_out;  // bf16 [B,L,D]

    char* ws = (char*)d_ws;
    size_t off = 0;
    float* gates = (float*)(ws + off);
    off += (size_t)B_DIM * E_NUM * sizeof(float);
    off = (off + 255) & ~(size_t)255;
    unsigned short* W1t = (unsigned short*)(ws + off);
    off += (size_t)E_NUM * H_DIM * D_DIM * sizeof(unsigned short);
    unsigned short* W2t = (unsigned short*)(ws + off);
    off += (size_t)E_NUM * D_DIM * H_DIM * sizeof(unsigned short);
    unsigned short* xb = (unsigned short*)(ws + off);

    gates_kernel<<<1, B_DIM, 0, stream>>>(logits, masks, gates);
    transpose_to_bf16<<<dim3(H_DIM / 32, D_DIM / 32, E_NUM), dim3(32, 8), 0,
                        stream>>>(W1, W1t, D_DIM, H_DIM);
    transpose_to_bf16<<<dim3(D_DIM / 32, H_DIM / 32, E_NUM), dim3(32, 8), 0,
                        stream>>>(W2, W2t, H_DIM, D_DIM);
    const int n4 = B_DIM * L_DIM * D_DIM / 4;
    convert_x_bf16<<<(n4 + 255) / 256, 256, 0, stream>>>(x, xb, n4);
    moe_ffn_kernel<<<dim3(B_DIM), 512, 0, stream>>>(xb, gates, W1t, W2t, b2, out);
}
